// CompoundGCN_38577396253198
// MI455X (gfx1250) — compile-verified, matrix operand order checked by reading, not run on hardware
//
#include <hip/hip_runtime.h>

#define N_NODES 50000
#define N_EDGES 600000
#define NODE_DIM 128
#define HIDDEN 256
#define OUT_DIM 16
#define N_GRAPHS 256
#define MAX_DEG 10
#define PERM_CAP 50176            // N_NODES + 11*16 pad, multiple of 16
#define TILES_M (PERM_CAP / 16)   // 3136
#define EDGE_TILES (N_EDGES / 16) // 37500 (exact)
#define INVALID_IDX 0xFFFFFFFFu
#define NEG_INF_KEY 0x007FFFFFu   // flip(bits(-inf))

typedef __attribute__((ext_vector_type(16))) __bf16 v16bf;
typedef __attribute__((ext_vector_type(8)))  __bf16 bf16x8;
typedef __attribute__((ext_vector_type(8)))  float  v8f;

// ---------------------------------------------------------------- helpers ---
// Fragment geometry (ISA 7.12.2, 16-bit A 16x32): lane L holds row m=L&15;
// its 16 elements are two contiguous K-runs of 8: [kk+8*kgrp, +8) and
// [kk+16+8*kgrp, +8), kgrp = L>>4.  B^T (n-major) mirrors this with n=L&15.

__device__ __forceinline__ void cvt8(const float4& a, const float4& b,
                                     v16bf& o, int base) {
  o[base + 0] = (__bf16)a.x; o[base + 1] = (__bf16)a.y;
  o[base + 2] = (__bf16)a.z; o[base + 3] = (__bf16)a.w;
  o[base + 4] = (__bf16)b.x; o[base + 5] = (__bf16)b.y;
  o[base + 6] = (__bf16)b.z; o[base + 7] = (__bf16)b.w;
}

__device__ __forceinline__ float4 relu4(float4 v) {
  v.x = fmaxf(v.x, 0.f); v.y = fmaxf(v.y, 0.f);
  v.z = fmaxf(v.z, 0.f); v.w = fmaxf(v.w, 0.f);
  return v;
}

// A fragment: 16 rows x 32 K of f32 activation data -> bf16 (4x b128 loads)
__device__ __forceinline__ v16bf load_a_row(const float* __restrict__ rp, int kk,
                                            int kgrp, int relu) {
  const float* q = rp + kk + 8 * kgrp;
  float4 a0 = *(const float4*)(q);
  float4 a1 = *(const float4*)(q + 4);
  float4 a2 = *(const float4*)(q + 16);
  float4 a3 = *(const float4*)(q + 20);
  if (relu) { a0 = relu4(a0); a1 = relu4(a1); a2 = relu4(a2); a3 = relu4(a3); }
  v16bf a;
  cvt8(a0, a1, a, 0);
  cvt8(a2, a3, a, 8);
  return a;
}

// B fragment from pre-transposed bf16 weights Wt[n][k] (n-major, ld = K):
// two contiguous 16-byte loads, zero conversions.
__device__ __forceinline__ v16bf load_bt_bf(const __bf16* __restrict__ Wt, int K,
                                            int kk, int n, int kgrp) {
  const __bf16* rp = Wt + (size_t)n * K + kk + 8 * kgrp;
  bf16x8 b0 = *(const bf16x8*)(rp);
  bf16x8 b1 = *(const bf16x8*)(rp + 16);
  v16bf b;
#pragma unroll
  for (int i = 0; i < 8; ++i) { b[i] = b0[i]; b[8 + i] = b1[i]; }
  return b;
}

__device__ __forceinline__ unsigned flip_f32(float f) {
  unsigned b = __float_as_uint(f);
  unsigned mask = (b & 0x80000000u) ? 0xFFFFFFFFu : 0x80000000u;
  return b ^ mask;
}

// ------------------------------------------------------------------ fills ---
__global__ void k_fill_u32(unsigned* p, unsigned v, size_t n) {
  size_t i = (size_t)blockIdx.x * blockDim.x + threadIdx.x;
  if (i < n) p[i] = v;
}

// ---------------------------------------- weight transpose+convert (once) ---
// W: nmat matrices [K x N] f32 (row-major) -> Wt: [N x K] bf16 (n-major)
__global__ void k_wcvt(const float* __restrict__ W, __bf16* __restrict__ Wt,
                       int K, int N, int nmat) {
  size_t idx = (size_t)blockIdx.x * blockDim.x + threadIdx.x;
  size_t per = (size_t)K * N;
  if (idx >= per * nmat) return;
  int m = (int)(idx / per);
  int r = (int)(idx - (size_t)m * per);
  int k = r / N, n = r - k * N;
  Wt[(size_t)m * per + (size_t)n * K + k] = (__bf16)W[idx];
}

// ------------------------------------------------------------ degree prep ---
__global__ void k_degree(const int* __restrict__ col, int* __restrict__ deg) {
  int e = blockIdx.x * blockDim.x + threadIdx.x;
  if (e < N_EDGES) atomicAdd(&deg[col[e]], 1);
}

__global__ void k_deg_clamp_count(int* __restrict__ deg, int* __restrict__ cnt) {
  int i = blockIdx.x * blockDim.x + threadIdx.x;
  if (i >= N_NODES) return;
  int d = deg[i]; if (d > MAX_DEG) d = MAX_DEG;
  deg[i] = d;
  atomicAdd(&cnt[d], 1);
}

__global__ void k_bucket_offsets(const int* __restrict__ cnt, int* __restrict__ poff,
                                 int* __restrict__ cur) {
  if (blockIdx.x == 0 && threadIdx.x == 0) {
    int off = 0;
    for (int d = 0; d <= MAX_DEG; ++d) {
      poff[d] = off; cur[d] = 0;
      off += (cnt[d] + 15) & ~15;   // 16-align each bucket
    }
    poff[MAX_DEG + 1] = off;
  }
}

__global__ void k_bucket_scatter(const int* __restrict__ deg, int* __restrict__ cur,
                                 const int* __restrict__ poff, unsigned* __restrict__ perm) {
  int i = blockIdx.x * blockDim.x + threadIdx.x;
  if (i >= N_NODES) return;
  int d = deg[i];
  int pos = atomicAdd(&cur[d], 1);
  perm[poff[d] + pos] = (unsigned)i;
}

// ---------------------------------------------------- MFConv: h = scatter ---
__global__ void k_scatter_h(const float* __restrict__ xin, const int* __restrict__ row,
                            const int* __restrict__ col, float* __restrict__ h, int relu) {
  size_t idx = (size_t)blockIdx.x * blockDim.x + threadIdx.x;  // E*32 threads
  int e = (int)(idx >> 5);
  if (e >= N_EDGES) return;
  int f = (int)(idx & 31) * 4;
  float4 v = *(const float4*)(xin + (size_t)row[e] * NODE_DIM + f);
  if (relu) v = relu4(v);
  float* dst = h + (size_t)col[e] * NODE_DIM + f;
  atomicAdd(dst + 0, v.x); atomicAdd(dst + 1, v.y);
  atomicAdd(dst + 2, v.z); atomicAdd(dst + 3, v.w);
}

// -------------------------------------------- MFConv degree-bucketed GEMM ---
// x_out[i] = h[i] @ W_lin[deg i] + b_lin[deg i] + relu?(x_in[i]) @ W_root[deg i]
__global__ void __launch_bounds__(32)
k_mfconv(const float* __restrict__ h, const float* __restrict__ xin, int relu_in,
         const __bf16* __restrict__ WlinT, const float* __restrict__ blin,
         const __bf16* __restrict__ WrootT, const int* __restrict__ deg,
         const unsigned* __restrict__ perm, float* __restrict__ xout) {
  int lane = threadIdx.x;
  int t = blockIdx.x;          // 16-node tile in padded perm order
  int n0 = blockIdx.y * 16;    // output-column tile
  unsigned node0 = perm[t * 16];
  if (node0 == INVALID_IDX) return;              // tile-uniform -> EXEC stays full
  int d = deg[node0];                            // bucket-uniform weight select
  int mA = lane & 15, kgrp = lane >> 4;
  unsigned nodeA = perm[t * 16 + mA];
  // padding lanes read node0's rows (harmless: their D rows are never stored)
  unsigned nodeLd = (nodeA != INVALID_IDX) ? nodeA : node0;
  const float* hp = h   + (size_t)nodeLd * NODE_DIM;
  const float* xp = xin + (size_t)nodeLd * NODE_DIM;
  int ncol = n0 + (lane & 15);
  const __bf16* WL = WlinT  + (size_t)d * NODE_DIM * NODE_DIM;
  const __bf16* WR = WrootT + (size_t)d * NODE_DIM * NODE_DIM;
  float bias = blin[d * NODE_DIM + ncol];
  v8f c;
#pragma unroll
  for (int r = 0; r < 8; ++r) c[r] = bias;
#pragma unroll
  for (int kk = 0; kk < NODE_DIM; kk += 32) {
    v16bf ah = load_a_row(hp, kk, kgrp, 0);
    v16bf bl = load_bt_bf(WL, NODE_DIM, kk, ncol, kgrp);
    c = __builtin_amdgcn_wmma_f32_16x16x32_bf16(false, ah, false, bl, (short)0, c, false, false);
    v16bf ax = load_a_row(xp, kk, kgrp, relu_in);
    v16bf br = load_bt_bf(WR, NODE_DIM, kk, ncol, kgrp);
    c = __builtin_amdgcn_wmma_f32_16x16x32_bf16(false, ax, false, br, (short)0, c, false, false);
  }
#pragma unroll
  for (int r = 0; r < 8; ++r) {
    unsigned nd = perm[t * 16 + r + 8 * kgrp];   // D row = r + 8*(lane>=16)
    if (nd != INVALID_IDX) xout[(size_t)nd * NODE_DIM + ncol] = c[r];
  }
}

// ------------------------------------------------ EdgeConv fused msg GEMM ---
// msg[e] = [xi, xj - xi] @ W_edge + b_edge ; atomic segment-max into acc[col[e]]
__global__ void __launch_bounds__(256)
k_edgeconv(const float* __restrict__ ein, int relu_in, const __bf16* __restrict__ WeT,
           const float* __restrict__ be, const int* __restrict__ row,
           const int* __restrict__ col, unsigned* __restrict__ acc) {
  __shared__ __align__(16) __bf16 Wt[128 * 256];   // pre-transposed bf16: Wt[n*256+k]
  {
    const float4* src = (const float4*)WeT;        // straight 64KB copy, b128 both ways
    float4* dst = (float4*)Wt;
    for (int i = threadIdx.x; i < 128 * 256 * 2 / 16; i += 256) dst[i] = src[i];
  }
  __syncthreads();
  int wave = threadIdx.x >> 5, lane = threadIdx.x & 31;
  int t = blockIdx.x * 8 + wave;                 // one wave per 16-edge tile
  if (t >= EDGE_TILES) return;                   // wave-uniform
  int mA = lane & 15, kgrp = lane >> 4;
  int e = t * 16 + mA;
  const float* xi = ein + (size_t)col[e] * 128;
  const float* xj = ein + (size_t)row[e] * 128;
  int nlane = lane & 15;
  v8f c[8];
#pragma unroll
  for (int nt = 0; nt < 8; ++nt) {
    float bias = be[nt * 16 + nlane];
#pragma unroll
    for (int r = 0; r < 8; ++r) c[nt][r] = bias;
  }
#pragma unroll
  for (int kk = 0; kk < 8; ++kk) {
    v16bf a;
    if (kk < 4) {                                // first half of K: xi
      a = load_a_row(xi, kk * 32, kgrp, relu_in);
    } else {                                     // second half: xj - xi
      const float* qi = xi + kk * 32 - 128 + 8 * kgrp;
      const float* qj = xj + kk * 32 - 128 + 8 * kgrp;
      float4 i0 = *(const float4*)(qi);
      float4 i1 = *(const float4*)(qi + 4);
      float4 i2 = *(const float4*)(qi + 16);
      float4 i3 = *(const float4*)(qi + 20);
      float4 j0 = *(const float4*)(qj);
      float4 j1 = *(const float4*)(qj + 4);
      float4 j2 = *(const float4*)(qj + 16);
      float4 j3 = *(const float4*)(qj + 20);
      if (relu_in) {
        i0 = relu4(i0); i1 = relu4(i1); i2 = relu4(i2); i3 = relu4(i3);
        j0 = relu4(j0); j1 = relu4(j1); j2 = relu4(j2); j3 = relu4(j3);
      }
      float4 d0, d1, d2, d3;
      d0.x = j0.x - i0.x; d0.y = j0.y - i0.y; d0.z = j0.z - i0.z; d0.w = j0.w - i0.w;
      d1.x = j1.x - i1.x; d1.y = j1.y - i1.y; d1.z = j1.z - i1.z; d1.w = j1.w - i1.w;
      d2.x = j2.x - i2.x; d2.y = j2.y - i2.y; d2.z = j2.z - i2.z; d2.w = j2.w - i2.w;
      d3.x = j3.x - i3.x; d3.y = j3.y - i3.y; d3.z = j3.z - i3.z; d3.w = j3.w - i3.w;
      cvt8(d0, d1, a, 0);
      cvt8(d2, d3, a, 8);
    }
#pragma unroll
    for (int nt = 0; nt < 8; ++nt) {
      // two ds_load_b128 per fragment
      v16bf b = load_bt_bf(Wt, 256, kk * 32, nt * 16 + nlane, kgrp);
      c[nt] = __builtin_amdgcn_wmma_f32_16x16x32_bf16(false, a, false, b, (short)0, c[nt], false, false);
    }
  }
  // segment-max scatter via order-preserving uint keys
#pragma unroll
  for (int nt = 0; nt < 8; ++nt) {
#pragma unroll
    for (int r = 0; r < 8; ++r) {
      int ee = t * 16 + r + 8 * kgrp;
      int seg = col[ee];
      atomicMax(&acc[(size_t)seg * 128 + nt * 16 + nlane], flip_f32(c[nt][r]));
    }
  }
}

__global__ void k_max_finalize(const unsigned* __restrict__ acc, float* __restrict__ eout, size_t n) {
  size_t i = (size_t)blockIdx.x * blockDim.x + threadIdx.x;
  if (i >= n) return;
  unsigned u = acc[i];
  unsigned b = (u & 0x80000000u) ? (u ^ 0x80000000u) : ~u;
  float f = __uint_as_float(b);
  if ((b & 0x7F800000u) == 0x7F800000u) f = 0.f;   // non-finite (empty segs) -> 0
  eout[i] = f;
}

// ------------------------------------------------------- readout scatter ----
__global__ void k_readout(const float* __restrict__ x2, const float* __restrict__ e2,
                          const int* __restrict__ row, const int* __restrict__ col,
                          const int* __restrict__ batch, float* __restrict__ pooled) {
  size_t idx = (size_t)blockIdx.x * blockDim.x + threadIdx.x;  // E*32 threads
  int e = (int)(idx >> 5);
  if (e >= N_EDGES) return;
  int f = (int)(idx & 31) * 4;
  int g = batch[col[e]];
  float4 xv = relu4(*(const float4*)(x2 + (size_t)row[e] * 128 + f));
  float4 ev = relu4(*(const float4*)(e2 + (size_t)col[e] * 128 + f));
  float* px = pooled + (size_t)g * 256 + f;
  float* pe = pooled + (size_t)g * 256 + 128 + f;
  atomicAdd(px + 0, xv.x); atomicAdd(px + 1, xv.y);
  atomicAdd(px + 2, xv.z); atomicAdd(px + 3, xv.w);
  atomicAdd(pe + 0, ev.x); atomicAdd(pe + 1, ev.y);
  atomicAdd(pe + 2, ev.z); atomicAdd(pe + 3, ev.w);
}

// ------------------------------------------------- generic GEMM+bias+relu ---
__global__ void __launch_bounds__(32)
k_gemm_bias_relu(const float* __restrict__ A, const __bf16* __restrict__ WT,
                 const float* __restrict__ bias, float* __restrict__ Out, int K, int N) {
  int lane = threadIdx.x;
  int m0 = blockIdx.x * 16, n0 = blockIdx.y * 16;
  int mA = lane & 15, kgrp = lane >> 4;
  const float* ap = A + (size_t)(m0 + mA) * K;
  int ncol = n0 + (lane & 15);
  float bv = bias[ncol];
  v8f c;
#pragma unroll
  for (int r = 0; r < 8; ++r) c[r] = bv;
  for (int kk = 0; kk < K; kk += 32) {
    v16bf a = load_a_row(ap, kk, kgrp, 0);
    v16bf b = load_bt_bf(WT, K, kk, ncol, kgrp);
    c = __builtin_amdgcn_wmma_f32_16x16x32_bf16(false, a, false, b, (short)0, c, false, false);
  }
#pragma unroll
  for (int r = 0; r < 8; ++r)
    Out[(size_t)(m0 + r + 8 * kgrp) * N + ncol] = fmaxf(c[r], 0.f);
}

// ------------------------------------------------------------------- host ---
static inline unsigned blks(size_t n, int b) { return (unsigned)((n + b - 1) / b); }

extern "C" void kernel_launch(void* const* d_in, const int* in_sizes, int n_in,
                              void* d_out, int out_size, void* d_ws, size_t ws_size,
                              hipStream_t stream) {
  const float* x    = (const float*)d_in[0];
  const float* ea   = (const float*)d_in[1];
  const int*   eidx = (const int*)d_in[2];
  const int*   row  = eidx;
  const int*   col  = eidx + N_EDGES;
  const int*   batch= (const int*)d_in[3];
  const float* Wlin = (const float*)d_in[4];
  const float* blin = (const float*)d_in[5];
  const float* Wroot= (const float*)d_in[6];
  const float* Wedge= (const float*)d_in[7];
  const float* bedge= (const float*)d_in[8];
  const float* W0   = (const float*)d_in[9];
  const float* b0   = (const float*)d_in[10];
  const float* W1   = (const float*)d_in[11];
  const float* b1   = (const float*)d_in[12];

  float* out      = (float*)d_out;                        // [256,16]
  float* emb_node = out + (size_t)N_GRAPHS * OUT_DIM;     // [50000,128]
  float* emb_edge = emb_node + (size_t)N_NODES * 128;     // [600000,128]

  char* p = (char*)d_ws;
  auto alloc = [&](size_t bytes) { char* r = p; p += (bytes + 511) & ~(size_t)511; return r; };
  int*      deg    = (int*)alloc((size_t)N_NODES * 4);
  int*      cnt    = (int*)alloc(64);
  int*      poff   = (int*)alloc(64);
  int*      cur    = (int*)alloc(64);
  unsigned* perm   = (unsigned*)alloc((size_t)PERM_CAP * 4);
  float*    hacc   = (float*)alloc((size_t)N_NODES * 128 * 4);  // h OR max-acc (reused)
  float*    bufX   = (float*)alloc((size_t)N_NODES * 128 * 4);  // x1 (pre-relu)
  float*    bufE   = (float*)alloc((size_t)N_NODES * 128 * 4);  // e1 (pre-relu)
  float*    pooled = (float*)alloc((size_t)N_GRAPHS * HIDDEN * 4);
  float*    hidden = (float*)alloc((size_t)N_GRAPHS * HIDDEN * 4);
  __bf16*   WlinT  = (__bf16*)alloc((size_t)(MAX_DEG + 1) * 128 * 128 * 2);
  __bf16*   WrootT = (__bf16*)alloc((size_t)(MAX_DEG + 1) * 128 * 128 * 2);
  __bf16*   WedgeT = (__bf16*)alloc((size_t)256 * 128 * 2);
  __bf16*   W0T    = (__bf16*)alloc((size_t)256 * 256 * 2);
  __bf16*   W1T    = (__bf16*)alloc((size_t)256 * 16 * 2);
  unsigned* acc    = (unsigned*)hacc;

  const size_t NF = (size_t)N_NODES * 128;

  // -------- one-time weight transpose+convert to bf16 (n-major)
  k_wcvt<<<blks((size_t)(MAX_DEG + 1) * 128 * 128, 256), 256, 0, stream>>>(Wlin, WlinT, 128, 128, MAX_DEG + 1);
  k_wcvt<<<blks((size_t)(MAX_DEG + 1) * 128 * 128, 256), 256, 0, stream>>>(Wroot, WrootT, 128, 128, MAX_DEG + 1);
  k_wcvt<<<blks((size_t)256 * 128, 256), 256, 0, stream>>>(Wedge, WedgeT, 256, 128, 1);
  k_wcvt<<<blks((size_t)256 * 256, 256), 256, 0, stream>>>(W0, W0T, 256, 256, 1);
  k_wcvt<<<blks((size_t)256 * 16, 256), 256, 0, stream>>>(W1, W1T, 256, 16, 1);

  // -------- degree + bucketed permutation (once)
  k_fill_u32<<<blks(N_NODES, 256), 256, 0, stream>>>((unsigned*)deg, 0u, N_NODES);
  k_fill_u32<<<1, 64, 0, stream>>>((unsigned*)cnt, 0u, 16);
  k_fill_u32<<<blks(PERM_CAP, 256), 256, 0, stream>>>(perm, INVALID_IDX, PERM_CAP);
  k_degree<<<blks(N_EDGES, 256), 256, 0, stream>>>(col, deg);
  k_deg_clamp_count<<<blks(N_NODES, 256), 256, 0, stream>>>(deg, cnt);
  k_bucket_offsets<<<1, 1, 0, stream>>>(cnt, poff, cur);
  k_bucket_scatter<<<blks(N_NODES, 256), 256, 0, stream>>>(deg, cur, poff, perm);

  // -------- two message-passing rounds
  for (int pass = 0; pass < 2; ++pass) {
    const float* xin  = pass ? bufX : x;
    const float* einp = pass ? bufE : ea;
    float*       xout = pass ? emb_node : bufX;
    float*       eout = pass ? emb_edge : bufE;
    int relu = pass ? 1 : 0;

    // MFConv
    k_fill_u32<<<blks(NF, 256), 256, 0, stream>>>((unsigned*)hacc, 0u, NF);
    k_scatter_h<<<blks((size_t)N_EDGES * 32, 256), 256, 0, stream>>>(xin, row, col, hacc, relu);
    k_mfconv<<<dim3(TILES_M, 8), 32, 0, stream>>>(hacc, xin, relu, WlinT, blin, WrootT,
                                                  deg, perm, xout);
    // EdgeConv (fused GEMM + segment-max)
    k_fill_u32<<<blks(NF, 256), 256, 0, stream>>>(acc, NEG_INF_KEY, NF);
    k_edgeconv<<<blks(EDGE_TILES, 8), 256, 0, stream>>>(einp, relu, WedgeT, bedge, row, col, acc);
    k_max_finalize<<<blks(NF, 256), 256, 0, stream>>>(acc, eout, NF);
  }

  // rows >= N_NODES of emb_edge are empty segments -> zero
  k_fill_u32<<<blks((size_t)(N_EDGES - N_NODES) * 128, 256), 256, 0, stream>>>(
      (unsigned*)(emb_edge + NF), 0u, (size_t)(N_EDGES - N_NODES) * 128);

  // -------- readout: edge scatter -> graph pool -> MLP
  k_fill_u32<<<blks((size_t)N_GRAPHS * HIDDEN, 256), 256, 0, stream>>>(
      (unsigned*)pooled, 0u, (size_t)N_GRAPHS * HIDDEN);
  k_readout<<<blks((size_t)N_EDGES * 32, 256), 256, 0, stream>>>(emb_node, emb_edge,
                                                                 row, col, batch, pooled);
  k_gemm_bias_relu<<<dim3(N_GRAPHS / 16, HIDDEN / 16), 32, 0, stream>>>(pooled, W0T, b0,
                                                                        hidden, HIDDEN, HIDDEN);
  k_gemm_bias_relu<<<dim3(N_GRAPHS / 16, OUT_DIM / 16), 32, 0, stream>>>(hidden, W1T, b1,
                                                                         out, HIDDEN, OUT_DIM);
}
